// NBEATSMoEBlock_58016418234528
// MI455X (gfx1250) — matrix operand order repaired and checked by reading, not compiled, not run on hardware
//
#include <hip/hip_runtime.h>
#include <hip/hip_bf16.h>

typedef __bf16 bf16_t;
typedef __attribute__((ext_vector_type(8)))  __bf16 v8bf;
typedef __attribute__((ext_vector_type(16))) __bf16 v16bf;
typedef __attribute__((ext_vector_type(8)))  float  v8f;

#define B_ROWS   8192
#define D_IN     512
#define D_HID    512
#define N_THETA  608
#define N_EXP    8
#define ROWS     64          // rows per chain-kernel tile (amortizes weight traffic)
#define LSTR     520         // padded LDS row stride (bf16 elems): 4-bank shift per row
#define MAXASG   (2*B_ROWS)  // total (row,expert) assignments
#define TILES_E  (MAXASG/ROWS)

static __device__ __forceinline__ v16bf cat16(v8bf a, v8bf b) {
    return __builtin_shufflevector(a, b, 0,1,2,3,4,5,6,7,8,9,10,11,12,13,14,15);
}
static __device__ __forceinline__ v16bf ldb(const bf16_t* bcol, int kk) {
    return cat16(*(const v8bf*)(bcol + kk), *(const v8bf*)(bcol + kk + 16));
}
static __device__ __forceinline__ void lda(const bf16_t* cur, int mrow, int khiA, int kk,
                                           v16bf (&a)[4]) {
    #pragma unroll
    for (int rt = 0; rt < 4; ++rt) {
        const bf16_t* ap = &cur[(rt * 16 + mrow) * LSTR + kk + khiA];
        a[rt] = cat16(*(const v8bf*)ap, *(const v8bf*)(ap + 16));
    }
}
#define WMMA_BF16(A, B, C) \
    __builtin_amdgcn_wmma_f32_16x16x32_bf16(false, (A), false, (B), (short)0, (C), false, false)

// async global->LDS copy: INST_OFFSET is added to BOTH the LDS and global address
#define ASYNC_CP(loff, ga, OFF)                                                \
    asm volatile("global_load_async_to_lds_b128 %0, %1, off offset:" #OFF      \
                 :: "v"(loff), "v"(ga) : "memory")

// ---------------- zero / init ----------------
__global__ void zero_f32_kernel(float* __restrict__ p, int n) {
    for (int i = blockIdx.x * blockDim.x + threadIdx.x; i < n; i += gridDim.x * blockDim.x)
        p[i] = 0.0f;
}
__global__ void init_counts_kernel(int* __restrict__ counts) {
    if (threadIdx.x < N_EXP) counts[threadIdx.x] = 0;
}

// ---------------- f32 -> bf16 (row-major copy) ----------------
__global__ void cvt_bf16_kernel(const float* __restrict__ src, bf16_t* __restrict__ dst, int n) {
    for (int i = blockIdx.x * blockDim.x + threadIdx.x; i < n; i += gridDim.x * blockDim.x)
        dst[i] = (bf16_t)src[i];
}

// ---------------- f32 (K,N) -> bf16 (N,K) transposed, batched ----------------
__global__ __launch_bounds__(256)
void transpose_cvt_kernel(const float* __restrict__ src, bf16_t* __restrict__ dst,
                          int K, int N) {
    __shared__ float tile[32][33];
    const int b  = blockIdx.z;
    const int n0 = blockIdx.x * 32, k0 = blockIdx.y * 32;
    const float* s = src + (size_t)b * K * N;
    bf16_t*      d = dst + (size_t)b * N * K;
    #pragma unroll
    for (int i = 0; i < 4; ++i) {
        int k = k0 + threadIdx.y + i * 8, n = n0 + threadIdx.x;
        if (k < K && n < N) tile[threadIdx.y + i * 8][threadIdx.x] = s[(size_t)k * N + n];
    }
    __syncthreads();
    #pragma unroll
    for (int i = 0; i < 4; ++i) {
        int n = n0 + threadIdx.y + i * 8, k = k0 + threadIdx.x;
        if (n < N && k < K) d[(size_t)n * K + k] = (bf16_t)tile[threadIdx.x][threadIdx.y + i * 8];
    }
}

// ---------------- gating: LN -> 8 logits -> top2 softmax ----------------
__global__ __launch_bounds__(256)
void gate_kernel(const float* __restrict__ x, const float* __restrict__ gamma,
                 const float* __restrict__ beta, const float* __restrict__ Wg,
                 int* __restrict__ topidx, float* __restrict__ topw,
                 int* __restrict__ counts) {
    __shared__ float sWg[D_IN * N_EXP];
    const int tid = threadIdx.x;
    for (int i = tid; i < D_IN * N_EXP; i += 256) sWg[i] = Wg[i];
    __syncthreads();

    const int wave = tid >> 5, lane = tid & 31;
    const int row  = blockIdx.x * 8 + wave;
    const float* xr = x + (size_t)row * D_IN;

    float xa[16];
    #pragma unroll
    for (int i = 0; i < 16; ++i) xa[i] = xr[lane * 16 + i];

    float s = 0.f, ss = 0.f;
    #pragma unroll
    for (int i = 0; i < 16; ++i) { s += xa[i]; ss += xa[i] * xa[i]; }
    #pragma unroll
    for (int m = 16; m >= 1; m >>= 1) { s += __shfl_xor(s, m, 32); ss += __shfl_xor(ss, m, 32); }
    const float mu   = s * (1.0f / D_IN);
    const float var  = ss * (1.0f / D_IN) - mu * mu;
    const float rstd = rsqrtf(var + 1e-5f);

    float xn[16];
    #pragma unroll
    for (int i = 0; i < 16; ++i) {
        int gi = lane * 16 + i;
        xn[i] = (xa[i] - mu) * rstd * gamma[gi] + beta[gi];
    }

    float lg[N_EXP];
    #pragma unroll
    for (int e = 0; e < N_EXP; ++e) {
        float p = 0.f;
        #pragma unroll
        for (int j = 0; j < 16; ++j) p += xn[j] * sWg[(lane * 16 + j) * N_EXP + e];
        #pragma unroll
        for (int m = 16; m >= 1; m >>= 1) p += __shfl_xor(p, m, 32);
        lg[e] = p;
    }

    if (lane == 0) {
        float v0 = -1e30f, v1 = -1e30f; int i0 = 0, i1 = 0;
        #pragma unroll
        for (int e = 0; e < N_EXP; ++e) {
            float v = lg[e];
            if (v > v0)      { v1 = v0; i1 = i0; v0 = v; i0 = e; }
            else if (v > v1) { v1 = v; i1 = e; }
        }
        const float e1 = expf(v1 - v0);
        const float w0 = 1.0f / (1.0f + e1);
        const float w1 = e1 * w0;
        topidx[row * 2 + 0] = i0; topidx[row * 2 + 1] = i1;
        topw[row * 2 + 0]   = w0; topw[row * 2 + 1]   = w1;
        atomicAdd(&counts[i0], 1); atomicAdd(&counts[i1], 1);
    }
}

// ---------------- prefix offsets ----------------
__global__ void offsets_kernel(const int* __restrict__ counts, int* __restrict__ offs,
                               int* __restrict__ cursors) {
    if (threadIdx.x == 0) {
        int a = 0;
        for (int e = 0; e < N_EXP; ++e) { offs[e] = a; a += counts[e]; cursors[e] = 0; }
    }
}

// ---------------- scatter rows into expert buckets ----------------
__global__ void scatter_kernel(const int* __restrict__ topidx, const float* __restrict__ topw,
                               const int* __restrict__ offs, int* __restrict__ cursors,
                               int* __restrict__ arow, float* __restrict__ aw) {
    const int r = blockIdx.x * blockDim.x + threadIdx.x;
    if (r >= B_ROWS) return;
    #pragma unroll
    for (int k = 0; k < 2; ++k) {
        int e = topidx[r * 2 + k];
        int p = atomicAdd(&cursors[e], 1);
        int sl = offs[e] + p;
        arow[sl] = r;
        aw[sl]   = topw[r * 2 + k];
    }
}

// ---------------- fused 5-layer expert chain (WMMA bf16, double-buffered LDS,
//                  hand software-pipelined k-loop, paired column tiles) ----------
__global__ __launch_bounds__(256)
void chain_kernel(const bf16_t* __restrict__ xb,   // [B,512] bf16 row-major
                  const bf16_t* __restrict__ W0t,  // [8][512out,512k] bf16 (transposed)
                  const bf16_t* __restrict__ Wmt,  // [3*8][512out,512k]
                  const bf16_t* __restrict__ Wot,  // [8][608out,512k]
                  const int*   __restrict__ counts,
                  const int*   __restrict__ offs,
                  const int*   __restrict__ arow,
                  const float* __restrict__ aw,
                  float*       __restrict__ out) {
    const int e    = blockIdx.y;
    const int cnt  = counts[e];
    const int base = blockIdx.x * ROWS;
    if (base >= cnt) return;
    const int off = offs[e];

    __shared__ bf16_t Abuf[2][ROWS * LSTR];   // activation ping-pong, ~130 KB
    __shared__ int    rid[ROWS];
    __shared__ float  rw[ROWS];

    const int tid  = threadIdx.x;
    const int wave = tid >> 5, lane = tid & 31;

    if (tid < ROWS) {
        int r = base + tid;
        if (r < cnt) { rid[tid] = arow[off + r]; rw[tid] = aw[off + r]; }
        else         { rid[tid] = 0;             rw[tid] = 0.0f; }
    }
    __syncthreads();

    { // async gather of 64 routed x-rows into LDS: 4 threads/row, 256B each
        const int r  = tid >> 2;
        const int c0 = (tid & 3) * 128;                 // bf16 elems, 256B per thread
        const uint64_t ga   = (uint64_t)(uintptr_t)(xb + (size_t)rid[r] * D_IN + c0);
        const uint32_t loff = (uint32_t)(uintptr_t)&Abuf[0][r * LSTR + c0];
        ASYNC_CP(loff, ga, 0);   ASYNC_CP(loff, ga, 16);
        ASYNC_CP(loff, ga, 32);  ASYNC_CP(loff, ga, 48);
        ASYNC_CP(loff, ga, 64);  ASYNC_CP(loff, ga, 80);
        ASYNC_CP(loff, ga, 96);  ASYNC_CP(loff, ga, 112);
        ASYNC_CP(loff, ga, 128); ASYNC_CP(loff, ga, 144);
        ASYNC_CP(loff, ga, 160); ASYNC_CP(loff, ga, 176);
        ASYNC_CP(loff, ga, 192); ASYNC_CP(loff, ga, 208);
        ASYNC_CP(loff, ga, 224); ASYNC_CP(loff, ga, 240);
        asm volatile("s_wait_asynccnt 0x0" ::: "memory");
    }
    __syncthreads();

    const int mrow = lane & 15;
    const int khiA = (lane >> 4) * 8;   // A-frag K offset per ISA 16-bit A layout
    const int khiB = (lane >> 4) * 8;   // B-frag K offset, same table order as A

    bf16_t* cur = &Abuf[0][0];
    bf16_t* nxt = &Abuf[1][0];

    // ---- layers 0..3 : N = 512, LDS(cur) -> LDS(nxt) ----
    for (int l = 0; l < 4; ++l) {
        const bf16_t* Wt = (l == 0)
            ? (W0t + (size_t)e * D_HID * D_IN)
            : (Wmt + ((size_t)(l - 1) * N_EXP + e) * D_HID * D_HID);

        for (int jj = 0; jj < 2; ++jj) {
            const int ctA = wave + jj * 8;
            const int ctB = ctA + 16;
            const bf16_t* bcolA = Wt + (size_t)(ctA * 16 + mrow) * 512 + khiB;
            const bf16_t* bcolB = Wt + (size_t)(ctB * 16 + mrow) * 512 + khiB;

            v8f acc[2][4] = {};
            v16bf b0A, b0B, b1A, b1B;
            v16bf a0[4], a1[4];
            // prologue: fragments for kk = 0
            b0A = ldb(bcolA, 0); b0B = ldb(bcolB, 0);
            lda(cur, mrow, khiA, 0, a0);
            for (int kk = 0; kk < 512; kk += 64) {
                // prefetch fragments for kk+32 while computing kk
                b1A = ldb(bcolA, kk + 32); b1B = ldb(bcolB, kk + 32);
                lda(cur, mrow, khiA, kk + 32, a1);
                #pragma unroll
                for (int rt = 0; rt < 4; ++rt) acc[0][rt] = WMMA_BF16(a0[rt], b0A, acc[0][rt]);
                #pragma unroll
                for (int rt = 0; rt < 4; ++rt) acc[1][rt] = WMMA_BF16(a0[rt], b0B, acc[1][rt]);
                // prefetch fragments for kk+64 while computing kk+32
                if (kk + 64 < 512) {
                    b0A = ldb(bcolA, kk + 64); b0B = ldb(bcolB, kk + 64);
                    lda(cur, mrow, khiA, kk + 64, a0);
                }
                #pragma unroll
                for (int rt = 0; rt < 4; ++rt) acc[0][rt] = WMMA_BF16(a1[rt], b1A, acc[0][rt]);
                #pragma unroll
                for (int rt = 0; rt < 4; ++rt) acc[1][rt] = WMMA_BF16(a1[rt], b1B, acc[1][rt]);
            }
            // write both column-tiles (disjoint LDS region per wave)
            #pragma unroll
            for (int p = 0; p < 2; ++p) {
                const int ct = p ? ctB : ctA;
                #pragma unroll
                for (int rt = 0; rt < 4; ++rt) {
                    #pragma unroll
                    for (int r = 0; r < 8; ++r) {
                        const int row = rt * 16 + khiA + r;  // C/D: lanes>=16 hold M+8
                        float v = acc[p][rt][r];
                        if (l > 0) v = fmaxf(v, 0.0f);       // ReLU only after Wmid layers
                        nxt[row * LSTR + ct * 16 + mrow] = (bf16_t)v;
                    }
                }
            }
        }
        __syncthreads();
        bf16_t* t = cur; cur = nxt; nxt = t;
    }

    // ---- layer 4 : Wout, N = 608 (38 col tiles), gate-scale + atomic scatter ----
    {
        const bf16_t* Wt = Wot + (size_t)e * N_THETA * D_IN;
        for (int j = 0; j < 5; ++j) {
            const int ct = wave + j * 8;
            if (ct >= 38) break;                         // wave-uniform
            const bf16_t* bcol = Wt + (size_t)(ct * 16 + mrow) * 512 + khiB;
            v8f acc[4] = {};
            v16bf b0, b1, a0[4], a1[4];
            b0 = ldb(bcol, 0);
            lda(cur, mrow, khiA, 0, a0);
            for (int kk = 0; kk < 512; kk += 64) {
                b1 = ldb(bcol, kk + 32);
                lda(cur, mrow, khiA, kk + 32, a1);
                #pragma unroll
                for (int rt = 0; rt < 4; ++rt) acc[rt] = WMMA_BF16(a0[rt], b0, acc[rt]);
                if (kk + 64 < 512) {
                    b0 = ldb(bcol, kk + 64);
                    lda(cur, mrow, khiA, kk + 64, a0);
                }
                #pragma unroll
                for (int rt = 0; rt < 4; ++rt) acc[rt] = WMMA_BF16(a1[rt], b1, acc[rt]);
            }
            const int c = ct * 16 + mrow;
            #pragma unroll
            for (int rt = 0; rt < 4; ++rt) {
                #pragma unroll
                for (int r = 0; r < 8; ++r) {
                    const int row = rt * 16 + khiA + r;
                    const float v = acc[rt][r] * rw[row];
                    const int rg  = rid[row];
                    if (c < D_IN)
                        atomicAdd(&out[(size_t)rg * D_IN + c], v);
                    else
                        atomicAdd(&out[(size_t)B_ROWS * D_IN + (size_t)rg * (N_THETA - D_IN) + (c - D_IN)], v);
                }
            }
        }
    }
}

// ---------------- host launcher ----------------
extern "C" void kernel_launch(void* const* d_in, const int* in_sizes, int n_in,
                              void* d_out, int out_size, void* d_ws, size_t ws_size,
                              hipStream_t stream) {
    const float* x     = (const float*)d_in[0];
    const float* gamma = (const float*)d_in[1];
    const float* beta  = (const float*)d_in[2];
    const float* Wg    = (const float*)d_in[3];
    const float* W0    = (const float*)d_in[4];
    const float* Wmid  = (const float*)d_in[5];
    const float* Wout  = (const float*)d_in[6];
    float* out = (float*)d_out;

    char* ws = (char*)d_ws;
    size_t o = 0;
    auto carve = [&](size_t bytes) -> void* {
        void* p = ws + o;
        o = (o + bytes + 255) & ~(size_t)255;
        return p;
    };
    bf16_t* xb   = (bf16_t*)carve((size_t)B_ROWS * D_IN * 2);
    bf16_t* W0t  = (bf16_t*)carve((size_t)N_EXP * D_HID * D_IN * 2);
    bf16_t* Wmt  = (bf16_t*)carve((size_t)3 * N_EXP * D_HID * D_HID * 2);
    bf16_t* Wot  = (bf16_t*)carve((size_t)N_EXP * N_THETA * D_IN * 2);
    int*    topidx  = (int*)carve((size_t)B_ROWS * 2 * 4);
    float*  topw    = (float*)carve((size_t)B_ROWS * 2 * 4);
    int*    counts  = (int*)carve(64 * 4);
    int*    offs    = (int*)carve(64 * 4);
    int*    cursors = (int*)carve(64 * 4);
    int*    arow    = (int*)carve((size_t)MAXASG * 4);
    float*  aw      = (float*)carve((size_t)MAXASG * 4);

    zero_f32_kernel<<<1024, 256, 0, stream>>>(out, B_ROWS * N_THETA);
    init_counts_kernel<<<1, 32, 0, stream>>>(counts);

    cvt_bf16_kernel<<<2048, 256, 0, stream>>>(x, xb, B_ROWS * D_IN);
    transpose_cvt_kernel<<<dim3(16, 16, N_EXP),     dim3(32, 8), 0, stream>>>(W0,   W0t, D_IN,  D_HID);
    transpose_cvt_kernel<<<dim3(16, 16, 3 * N_EXP), dim3(32, 8), 0, stream>>>(Wmid, Wmt, D_HID, D_HID);
    transpose_cvt_kernel<<<dim3(19, 16, N_EXP),     dim3(32, 8), 0, stream>>>(Wout, Wot, D_IN,  N_THETA);

    gate_kernel<<<B_ROWS / 8, 256, 0, stream>>>(x, gamma, beta, Wg, topidx, topw, counts);
    offsets_kernel<<<1, 1, 0, stream>>>(counts, offs, cursors);
    scatter_kernel<<<B_ROWS / 256, 256, 0, stream>>>(topidx, topw, offs, cursors, arow, aw);

    chain_kernel<<<dim3(TILES_E, N_EXP), 256, 0, stream>>>(
        xb, W0t, Wmt, Wot, counts, offs, arow, aw, out);
}
